// ModalGCN_10900626998075
// MI455X (gfx1250) — matrix operand order repaired and checked by reading, not compiled, not run on hardware
//
#include <hip/hip_runtime.h>
#include <hip/hip_bf16.h>

// ---------------------------------------------------------------------------
// ModalGCN on MI455X (gfx1250, wave32).
// Scatter-bound workload (2M edges x 64 dims of gather + f32 atomics, L2
// resident); dense parts use V_WMMA_F32_16X16X32_BF16 with f32 accumulate.
// Weight (B) operand is staged into LDS once per workgroup with the Tensor
// Data Mover (tensor_load_to_lds, TENSORcnt), so the global-memory pipe only
// streams the A operand.
// ---------------------------------------------------------------------------

#define DIM_LATENT 256
#define DIM_ID     64

typedef __bf16   bf16_t;
typedef bf16_t   v16bf __attribute__((ext_vector_type(16)));
typedef float    v8f   __attribute__((ext_vector_type(8)));
typedef unsigned short us8  __attribute__((ext_vector_type(8)));
typedef unsigned short us16 __attribute__((ext_vector_type(16)));
typedef int      si4 __attribute__((ext_vector_type(4)));
typedef int      si8 __attribute__((ext_vector_type(8)));

__device__ __forceinline__ unsigned short f2bf(float f) {
  unsigned int u = __float_as_uint(f);
  u += 0x7FFFu + ((u >> 16) & 1u);   // round-to-nearest-even
  return (unsigned short)(u >> 16);
}
__device__ __forceinline__ float lrelu(float v) { return v > 0.0f ? v : 0.01f * v; }

// TDM: DMA a 2D tensor of 2-byte elements [dim1 rows x dim0 elems], row
// stride stride0 elems, from global into LDS at lds_off. D# per ISA 8.3/8.4.
__device__ __forceinline__ void tdm_load_2d_to_lds(const void* gptr, unsigned lds_off,
                                                   int dim0, int dim1, int stride0) {
  unsigned long long ga = (unsigned long long)(uintptr_t)gptr;
  si4 g0;
  g0[0] = 1;                                            // count=1 (valid user D#)
  g0[1] = (int)lds_off;                                 // lds_addr (bytes)
  g0[2] = (int)(ga & 0xFFFFFFFFull);                    // global_addr[31:0]
  g0[3] = (int)((ga >> 32) & 0x1FFFFFFull) | (2 << 30); // global_addr[56:32], type=2
  si8 g1;
  g1[0] = 1 << 16;                                      // data_size=1 (2 bytes)
  g1[1] = (dim0 & 0xFFFF) << 16;                        // tensor_dim0[15:0] @ bits 63:48
  g1[2] = ((dim0 >> 16) & 0xFFFF) | ((dim1 & 0xFFFF) << 16);  // dim0[31:16] | dim1[15:0]
  g1[3] = ((dim1 >> 16) & 0xFFFF) | ((dim0 & 0xFFFF) << 16);  // dim1[31:16] | tile_dim0=dim0
  g1[4] = dim1 & 0xFFFF;                                // tile_dim1=dim1, tile_dim2=0
  g1[5] = stride0;                                      // tensor_dim0_stride[31:0]
  g1[6] = 0;                                            // stride0[47:32]=0, dim1_stride lo=0
  g1[7] = 0;
  asm volatile("tensor_load_to_lds %0, %1" :: "s"(g0), "s"(g1) : "memory");
}

// ---------------- elementwise / graph kernels ------------------------------

__global__ void k_zero(float* __restrict__ p, int n) {
  int i = blockIdx.x * blockDim.x + threadIdx.x;
  if (i < n) p[i] = 0.0f;
}

__global__ void k_deg(const int* __restrict__ row, float* __restrict__ deg, int E) {
  int e = blockIdx.x * blockDim.x + threadIdx.x;
  if (e < E) atomicAdd(&deg[row[e]], 1.0f);
}

__global__ void k_dis(float* __restrict__ deg, int n) {   // in place: deg -> deg^-0.5
  int i = blockIdx.x * blockDim.x + threadIdx.x;
  if (i < n) deg[i] = rsqrtf(fmaxf(deg[i], 1.0f));
}

__global__ void k_edge_norm(const int* __restrict__ ei, const float* __restrict__ dis,
                            float* __restrict__ nrm, int E) {
  int e = blockIdx.x * blockDim.x + threadIdx.x;
  if (e < E) nrm[e] = dis[ei[e]] * dis[ei[E + e]];
}

// concat(preference, features) -> L2-row-normalize -> bf16 A matrix [N, 256]
__global__ void k_rows(const float* __restrict__ pref, const float* __restrict__ feat,
                       unsigned short* __restrict__ xbf, int num_user) {
  __shared__ float red[DIM_LATENT];
  int r = blockIdx.x, t = threadIdx.x;
  const float* src = (r < num_user) ? (pref + (size_t)r * DIM_LATENT)
                                    : (feat + (size_t)(r - num_user) * DIM_LATENT);
  float v = src[t];
  red[t] = v * v;
  __syncthreads();
  for (int s = DIM_LATENT / 2; s > 0; s >>= 1) {
    if (t < s) red[t] += red[t + s];
    __syncthreads();
  }
  float scale = 1.0f / fmaxf(sqrtf(red[0]), 1e-12f);
  xbf[(size_t)r * DIM_LATENT + t] = f2bf(v * scale);
}

// W[K, 64] (row-major, f32) -> out[(coloff+n)*K + kk] = bf16(W[kk,n])  (B^T)
__global__ void k_wt(const float* __restrict__ W, unsigned short* __restrict__ out,
                     int K, int coloff) {
  int tid = blockIdx.x * blockDim.x + threadIdx.x;
  if (tid >= 64 * K) return;
  int n = tid / K, kk = tid % K;
  out[(size_t)(coloff + n) * K + kk] = f2bf(W[(size_t)kk * 64 + n]);
}

// h[row] += hW[col] * norm[e]  (16 threads/edge, 4 dims each, f32 atomics in L2)
__global__ void k_scatter(const int* __restrict__ ei, const float* __restrict__ nrm,
                          const float* __restrict__ c2, float* __restrict__ h, int E) {
  int tid = blockIdx.x * blockDim.x + threadIdx.x;
  if (tid >= E * 16) return;
  int e  = tid >> 4;
  int dg = (tid & 15) << 2;
  int r = ei[e], c = ei[E + e];
  float nv = nrm[e];
  const float4 v = *(const float4*)(c2 + (size_t)c * 128 + dg);
  float* hp = h + (size_t)r * 64 + dg;
  atomicAdd(hp + 0, v.x * nv);
  atomicAdd(hp + 1, v.y * nv);
  atomicAdd(hp + 2, v.z * nv);
  atomicAdd(hp + 3, v.w * nv);
}

__global__ void k_hact(const float* __restrict__ h, unsigned short* __restrict__ hbf, int n) {
  int i = blockIdx.x * blockDim.x + threadIdx.x;
  if (i < n) hbf[i] = f2bf(lrelu(h[i]));
}

// x = lrelu(g + g_b + lrelu(xhat_pre + lin_b) + id_emb); emit f32 and/or bf16
__global__ void k_fuse(const float* __restrict__ g, const float* __restrict__ gb,
                       const float* __restrict__ c2, const float* __restrict__ lb,
                       const float* __restrict__ id_emb,
                       float* __restrict__ xout_f32, unsigned short* __restrict__ xout_bf,
                       int total) {
  int i = blockIdx.x * blockDim.x + threadIdx.x;
  if (i >= total) return;
  int d = i & 63, node = i >> 6;
  float xh = lrelu(c2[(size_t)node * 128 + 64 + d] + lb[d]) + id_emb[i];
  float v  = lrelu(g[i] + gb[d] + xh);
  if (xout_f32) xout_f32[i] = v;
  if (xout_bf)  xout_bf[i]  = f2bf(v);
}

// ---------------- WMMA GEMM: C[M,Nout] = A[M,K](bf16) x B[K,Nout](bf16^T) ---
// One wave per 16x16 tile; blockDim.x = 32*(Nout/16) waves cover the N dim.
// The whole Bt [Nout, K] weight tile is DMA'd into LDS once per workgroup by
// the Tensor Data Mover; the k-loop then streams A from global (b128 loads)
// and reads B fragments from LDS (ds_load_b128).
// A fragment (16x32 bf16, ISA 7.12.2): lanes 0-15 row m=lane, K {0..7,16..23};
// lanes 16-31 row m=lane-16, K {8..15,24..31}. B fragment from Bt[n, :]:
// lanes 0-15 col n=lane K {0..7,16..23}; lanes 16-31 col n=lane-16 K {8..15,24..31}.
__global__ void k_gemm_bf16(const unsigned short* __restrict__ A,
                            const unsigned short* __restrict__ Bt,
                            float* __restrict__ C, int K, int Nout) {
  extern __shared__ unsigned short smemB[];   // [Nout, K] bf16

  if (threadIdx.x < 32) {                     // wave 0 issues the DMA
    tdm_load_2d_to_lds(Bt, (unsigned)(uintptr_t)smemB, K, Nout, K);
    __builtin_amdgcn_s_wait_tensorcnt(0);
  }
  __syncthreads();

  const int lane = threadIdx.x & 31;
  const int wave = threadIdx.x >> 5;         // n-tile index
  const int half = lane >> 4;
  const int m    = lane & 15;
  const int nloc = lane & 15;

  const unsigned short* arow = A + (size_t)(blockIdx.x * 16 + m) * K;
  const unsigned short* brow = smemB + (size_t)(wave * 16 + nloc) * K;

  v8f acc = {};
  for (int k = 0; k < K; k += 32) {
    __builtin_prefetch(arow + k + 64, 0, 1);   // global_prefetch_b8
    us8 a0 = *(const us8*)(arow + k + half * 8);
    us8 a1 = *(const us8*)(arow + k + 16 + half * 8);
    us8 b0 = *(const us8*)(brow + k + half * 8);
    us8 b1 = *(const us8*)(brow + k + 16 + half * 8);
    us16 au = __builtin_shufflevector(a0, a1, 0,1,2,3,4,5,6,7,8,9,10,11,12,13,14,15);
    us16 bu = __builtin_shufflevector(b0, b1, 0,1,2,3,4,5,6,7,8,9,10,11,12,13,14,15);
    v16bf av = __builtin_bit_cast(v16bf, au);
    v16bf bv = __builtin_bit_cast(v16bf, bu);
    acc = __builtin_amdgcn_wmma_f32_16x16x32_bf16(
        /*neg_a=*/false, av, /*neg_b=*/false, bv,
        /*c_mod=*/(short)0, acc, /*reuse_a=*/false, /*reuse_b=*/false);
  }

  // C/D layout: lanes 0-15 -> N=lane, M=r; lanes 16-31 -> N=lane-16, M=r+8
  const int colg  = wave * 16 + nloc;
  const int mbase = blockIdx.x * 16 + half * 8;
  float* cp = C + (size_t)mbase * Nout + colg;
#pragma unroll
  for (int r = 0; r < 8; ++r) cp[(size_t)r * Nout] = acc[r];
}

// ---------------------------------------------------------------------------

static inline int cdiv(int a, int b) { return (a + b - 1) / b; }

extern "C" void kernel_launch(void* const* d_in, const int* in_sizes, int n_in,
                              void* d_out, int out_size, void* d_ws, size_t ws_size,
                              hipStream_t stream) {
  const float* features = (const float*)d_in[0];
  const float* id_emb   = (const float*)d_in[1];
  const float* pref     = (const float*)d_in[2];
  const float* W_gcn0   = (const float*)d_in[3];
  const float* W_gcn1   = (const float*)d_in[4];
  const float* lin_w0   = (const float*)d_in[5];
  const float* lin_b0   = (const float*)d_in[6];
  const float* lin_w1   = (const float*)d_in[7];
  const float* lin_b1   = (const float*)d_in[8];
  const float* g_w0     = (const float*)d_in[9];
  const float* g_b0     = (const float*)d_in[10];
  const float* g_w1     = (const float*)d_in[11];
  const float* g_b1     = (const float*)d_in[12];
  const int*   ei       = (const int*)d_in[13];

  const int NUM_ITEM  = in_sizes[0] / DIM_LATENT;     // 50000
  const int NUM_NODES = in_sizes[1] / DIM_ID;         // 100000
  const int NUM_USER  = NUM_NODES - NUM_ITEM;         // 50000
  const int E         = in_sizes[13] / 2;             // 2,000,000
  const int ND        = NUM_NODES * DIM_ID;           // 6.4M

  // ---- workspace carve-out ----
  char* ws = (char*)d_ws;
  size_t off = 0;
  auto carve = [&](size_t bytes) -> void* {
    void* p = ws + off;
    off = (off + bytes + 255) & ~(size_t)255;
    return p;
  };
  float*          dis  = (float*)         carve((size_t)NUM_NODES * 4);
  float*          nrm  = (float*)         carve((size_t)E * 4);
  unsigned short* xbf  = (unsigned short*)carve((size_t)NUM_NODES * DIM_LATENT * 2);
  float*          c2   = (float*)         carve((size_t)NUM_NODES * 128 * 4);  // [hW | xhat_pre]
  float*          h    = (float*)         carve((size_t)ND * 4);
  unsigned short* hbf  = (unsigned short*)carve((size_t)ND * 2);
  float*          gbuf = (float*)         carve((size_t)ND * 4);
  unsigned short* wcat = (unsigned short*)carve((size_t)128 * DIM_LATENT * 2); // [128, K]^T
  unsigned short* gwt  = (unsigned short*)carve((size_t)64 * 64 * 2);

  const int TB = 256;
  const int mtiles = NUM_NODES / 16;   // 6250

  // ---- degree norm ----
  k_zero     <<<cdiv(NUM_NODES, TB), TB, 0, stream>>>(dis, NUM_NODES);
  k_deg      <<<cdiv(E, TB),         TB, 0, stream>>>(ei, dis, E);
  k_dis      <<<cdiv(NUM_NODES, TB), TB, 0, stream>>>(dis, NUM_NODES);
  k_edge_norm<<<cdiv(E, TB),         TB, 0, stream>>>(ei, dis, nrm, E);

  // ---- x = rownorm(concat(pref, feat)) -> bf16 [N,256] ----
  k_rows<<<NUM_NODES, DIM_LATENT, 0, stream>>>(pref, features, xbf, NUM_USER);

  // =================== layer 1 (K = 256) ===================
  {
    const int K = DIM_LATENT;
    k_wt<<<cdiv(64 * K, TB), TB, 0, stream>>>(W_gcn0, wcat, K, 0);
    k_wt<<<cdiv(64 * K, TB), TB, 0, stream>>>(lin_w0, wcat, K, 64);
    k_wt<<<cdiv(64 * 64, TB), TB, 0, stream>>>(g_w0, gwt, 64, 0);

    k_gemm_bf16<<<mtiles, 256, (size_t)128 * K * 2, stream>>>(xbf, wcat, c2, K, 128);

    k_zero   <<<cdiv(ND, TB), TB, 0, stream>>>(h, ND);
    k_scatter<<<cdiv(E * 16, TB), TB, 0, stream>>>(ei, nrm, c2, h, E);
    k_hact   <<<cdiv(ND, TB), TB, 0, stream>>>(h, hbf, ND);

    k_gemm_bf16<<<mtiles, 128, (size_t)64 * 64 * 2, stream>>>(hbf, gwt, gbuf, 64, 64);

    // next-layer bf16 input overwrites the front of xbf ([N,64])
    k_fuse<<<cdiv(ND, TB), TB, 0, stream>>>(gbuf, g_b0, c2, lin_b0, id_emb,
                                            nullptr, xbf, ND);
  }

  // =================== layer 2 (K = 64) ====================
  {
    const int K = DIM_ID;
    k_wt<<<cdiv(64 * K, TB), TB, 0, stream>>>(W_gcn1, wcat, K, 0);
    k_wt<<<cdiv(64 * K, TB), TB, 0, stream>>>(lin_w1, wcat, K, 64);
    k_wt<<<cdiv(64 * 64, TB), TB, 0, stream>>>(g_w1, gwt, 64, 0);

    k_gemm_bf16<<<mtiles, 256, (size_t)128 * K * 2, stream>>>(xbf, wcat, c2, K, 128);

    k_zero   <<<cdiv(ND, TB), TB, 0, stream>>>(h, ND);
    k_scatter<<<cdiv(E * 16, TB), TB, 0, stream>>>(ei, nrm, c2, h, E);
    k_hact   <<<cdiv(ND, TB), TB, 0, stream>>>(h, hbf, ND);

    k_gemm_bf16<<<mtiles, 128, (size_t)64 * 64 * 2, stream>>>(hbf, gwt, gbuf, 64, 64);

    k_fuse<<<cdiv(ND, TB), TB, 0, stream>>>(gbuf, g_b1, c2, lin_b1, id_emb,
                                            (float*)d_out, nullptr, ND);
  }
}
